// DiffViT_CLS_61280593380060
// MI455X (gfx1250) — compile-verified, matrix operand order checked by reading, not run on hardware
//
#include <hip/hip_runtime.h>
#include <hip/hip_bf16.h>
#include <math.h>

typedef __bf16 bf16;
typedef __attribute__((ext_vector_type(16))) __bf16 v16bf;
typedef __attribute__((ext_vector_type(8)))  __bf16 v8bf;
typedef __attribute__((ext_vector_type(8)))  float  v8f;
typedef __attribute__((ext_vector_type(4)))  unsigned u32x4;
typedef __attribute__((ext_vector_type(8)))  unsigned u32x8;

#define CHN 64
#define IMGSZ 64
#define EMB 192
#define NCLS 16
#define BATCH 16
#define DSTATE 16
#define DINNER 384
#define DTRANK 12
#define SEQL 1024
#define HW 4096
#define NCOL (BATCH*HW)   /* 65536 */
#define NROW (BATCH*SEQL) /* 16384 */
#define PDIM 256
#define KMAX 576
#define EPSV 1e-5f

__device__ __forceinline__ bf16 f2bf(float f) {
  unsigned u = __builtin_bit_cast(unsigned, f);
  unsigned r = (u + 0x7FFFu + ((u >> 16) & 1u)) >> 16;
  return __builtin_bit_cast(bf16, (unsigned short)r);
}
__device__ __forceinline__ float siluf(float x) { return x / (1.f + __expf(-x)); }

// ---------------------------------------------------------------------------
// bf16 WMMA GEMM with TDM-staged A operand.
// C[M x N] = A[M x K] * B, with B given transposed as Bt[N x K] row-major.
// One wave (32 lanes) per workgroup computes a 32x16 C macro-tile as two
// 16x16 WMMA tiles sharing the B operand stream (B traffic halved).
// The 32xK slab of A is DMA'd into LDS by the Tensor Data Mover
// (tensor_load_to_lds, 2-D D#, TENSORcnt), then consumed via ds_load_b128.
// 16-bit A/B VGPR striping per CDNA5 ISA:
//   lanes 0-15 : row/col = lane,    K = {k0+0..7,  k0+16..23}
//   lanes 16-31: row/col = lane-16, K = {k0+8..15, k0+24..31}
// ---------------------------------------------------------------------------
__global__ void gemm_bf16_wmma(const bf16* __restrict__ A, const bf16* __restrict__ Bt,
                               float* __restrict__ C, int M, int N, int K) {
  __shared__ __attribute__((aligned(64))) bf16 As[32 * KMAX];

  const int tm   = blockIdx.x * 32;
  const int tn   = blockIdx.y * 16;
  const int lane = threadIdx.x;
  const int half = lane >> 4;
  const int l15  = lane & 15;
  const int kb   = half * 8;

  // ---- Tensor DMA Descriptor (D#): 2-D tensor, data_size=2B, tile K x 32 ----
  unsigned lds = (unsigned)(unsigned long long)(As);
  unsigned long long ga = (unsigned long long)(A + (size_t)tm * K);
  u32x4 g0 = { 1u,                                    // count=1 (valid user D#)
               lds,                                   // lds_addr
               (unsigned)ga,                          // global_addr[31:0]
               (unsigned)((ga >> 32) & 0x01FFFFFFu) | 0x80000000u }; // addr[56:32] | type=2
  u32x8 g1 = { 0x00010000u,                           // mask=0, data_size=1 (2 bytes)
               ((unsigned)K << 16),                   // tensor_dim0[15:0]
               (32u << 16),                           // dim0 hi=0 | tensor_dim1=32
               ((unsigned)K << 16),                   // dim1 hi=0 | tile_dim0=K
               32u,                                   // tile_dim1=32, tile_dim2=0
               (unsigned)K,                           // tensor_dim0_stride[31:0]
               0u, 0u };
  asm volatile("tensor_load_to_lds %0, %1" :: "s"(g0), "s"(g1) : "memory");
  asm volatile("s_wait_tensorcnt 0x0" ::: "memory");

  const long brow = (long)(tn + l15) * K;
  const bf16* a0p = As + l15 * K;
  const bf16* a1p = As + (16 + l15) * K;
  v8f acc0 = {};
  v8f acc1 = {};
  for (int k0 = 0; k0 < K; k0 += 32) {
    v8bf blo = *reinterpret_cast<const v8bf*>(Bt + brow + k0 + kb);
    v8bf bhi = *reinterpret_cast<const v8bf*>(Bt + brow + k0 + kb + 16);
    v16bf b = __builtin_shufflevector(blo, bhi, 0,1,2,3,4,5,6,7,8,9,10,11,12,13,14,15);
    v8bf a0lo = *reinterpret_cast<const v8bf*>(a0p + k0 + kb);
    v8bf a0hi = *reinterpret_cast<const v8bf*>(a0p + k0 + kb + 16);
    v8bf a1lo = *reinterpret_cast<const v8bf*>(a1p + k0 + kb);
    v8bf a1hi = *reinterpret_cast<const v8bf*>(a1p + k0 + kb + 16);
    v16bf a0 = __builtin_shufflevector(a0lo, a0hi, 0,1,2,3,4,5,6,7,8,9,10,11,12,13,14,15);
    v16bf a1 = __builtin_shufflevector(a1lo, a1hi, 0,1,2,3,4,5,6,7,8,9,10,11,12,13,14,15);
    acc0 = __builtin_amdgcn_wmma_f32_16x16x32_bf16(false, a0, false, b, (short)0, acc0,
                                                   false, false);
    acc1 = __builtin_amdgcn_wmma_f32_16x16x32_bf16(false, a1, false, b, (short)0, acc1,
                                                   false, false);
  }
#pragma unroll
  for (int r = 0; r < 8; ++r) {
    C[(long)(tm + r + half * 8) * N + tn + l15]      = acc0[r];
    C[(long)(tm + 16 + r + half * 8) * N + tn + l15] = acc1[r];
  }
}

// --------------------------- elementwise kernels ---------------------------

__global__ void k_zero1(float* p) { *p = 0.f; }

__global__ void k_cast_bf16(const float* __restrict__ in, bf16* __restrict__ out, int n) {
  int i = blockIdx.x * 256 + threadIdx.x;
  if (i < n) out[i] = f2bf(in[i]);
}

// Bt for conv1: bt[(b*4096+hw)*64 + c] = x[b,c,hw]
__global__ void k_pack_x_bt(const float* __restrict__ x, bf16* __restrict__ bt) {
  int i = blockIdx.x * 256 + threadIdx.x;      // 64*65536
  int c = i & 63; int n = i >> 6;
  int b = n >> 12; int hw = n & 4095;
  bt[i] = f2bf(x[((b << 6) + c) * 4096 + hw]);
}

// nd = (x0 - center)^2, global sum via block reduce + atomic
__global__ void k_fuzzy_sum(const float* __restrict__ x0, float* __restrict__ sum) {
  int i = blockIdx.x * 256 + threadIdx.x;      // 192*65536
  int n = i & 65535; int e = i >> 16; int b = n >> 12;
  float c = x0[(e << 16) + (b << 12) + 2080];  // h=32,w=32
  float d = x0[i] - c;
  __shared__ float red[256];
  red[threadIdx.x] = d * d;
  __syncthreads();
  for (int s = 128; s > 0; s >>= 1) {
    if (threadIdx.x < s) red[threadIdx.x] += red[threadIdx.x + s];
    __syncthreads();
  }
  if (threadIdx.x == 0) atomicAdd(sum, red[0]);
}

__global__ void k_fuzzy_w(const float* __restrict__ x0, const float* __restrict__ sum,
                          float* __restrict__ wfz) {
  int i = blockIdx.x * 256 + threadIdx.x;
  int n = i & 65535; int e = i >> 16; int b = n >> 12;
  float c = x0[(e << 16) + (b << 12) + 2080];
  float d = x0[i] - c;
  float nd = d * d;
  int hw = n & 4095; int hi = hw >> 6; int wi = hw & 63;
  int ai = hi > 31 ? hi - 31 : 31 - hi;
  int aj = wi > 31 ? wi - 31 : 31 - wi;
  float wm = __expf(-(float)(ai + aj));        // LAMBDA_M = 1
  wfz[i] = __expf(-nd / sum[0]) * wm;          // LAMBDA_G = 1
}

// x1 = dwconv3x3(relu(bn2(x0))), pad=1
__global__ void k_bnrelu_dwconv(const float* __restrict__ x0, const float* __restrict__ g,
                                const float* __restrict__ bq, const float* __restrict__ rm,
                                const float* __restrict__ rv, const float* __restrict__ w2,
                                float* __restrict__ x1) {
  int i = blockIdx.x * 256 + threadIdx.x;      // 192*65536
  int n = i & 65535; int e = i >> 16;
  int b = n >> 12; int hw = n & 4095; int h = hw >> 6; int w = hw & 63;
  float sc = g[e] * rsqrtf(rv[e] + EPSV);
  float sh = bq[e] - rm[e] * sc;
  float acc = 0.f;
#pragma unroll
  for (int dy = 0; dy < 3; ++dy)
#pragma unroll
    for (int dx = 0; dx < 3; ++dx) {
      int hh = h + dy - 1, ww = w + dx - 1;
      if (hh >= 0 && hh < 64 && ww >= 0 && ww < 64) {
        float v = x0[(e << 16) + (b << 12) + hh * 64 + ww] * sc + sh;
        v = fmaxf(v, 0.f);
        acc += w2[e * 9 + dy * 3 + dx] * v;
      }
    }
  x1[i] = acc;
}

// x2 = relu(bn3(concat[w*x0, x1, w*x0])) packed as Bt [65536 x 576] bf16
__global__ void k_build_x2(const float* __restrict__ x0, const float* __restrict__ wfz,
                           const float* __restrict__ x1, const float* __restrict__ g,
                           const float* __restrict__ bq, const float* __restrict__ rm,
                           const float* __restrict__ rv, bf16* __restrict__ x2t) {
  int i = blockIdx.x * 256 + threadIdx.x;      // 576*65536
  int n = i & 65535; int cc = i >> 16;
  float v;
  if (cc < 192)       v = wfz[(cc << 16) + n] * x0[(cc << 16) + n];
  else if (cc < 384)  v = x1[((cc - 192) << 16) + n];
  else                v = wfz[((cc - 384) << 16) + n] * x0[((cc - 384) << 16) + n];
  float sc = g[cc] * rsqrtf(rv[cc] + EPSV);
  v = (v - rm[cc]) * sc + bq[cc];
  x2t[(long)n * 576 + cc] = f2bf(fmaxf(v, 0.f));
}

__global__ void k_residual(const float* __restrict__ c3, const float* __restrict__ x,
                           float* __restrict__ xf) {
  int i = blockIdx.x * 256 + threadIdx.x;      // 16*64*4096
  int hw = i & 4095; int bc = i >> 12; int c = bc & 63; int b = bc >> 6;
  xf[i] = c3[(c << 16) + (b << 12) + hw] + x[i];
}

// patches[(b*1024+p)*256 + k], k = c*4 + py*2 + px
__global__ void k_pack_patches(const float* __restrict__ xf, bf16* __restrict__ pat) {
  int i = blockIdx.x * 256 + threadIdx.x;      // 16384*256
  int k = i & 255; int row = i >> 8;
  int b = row >> 10; int p = row & 1023;
  int c = k >> 2; int py = (k >> 1) & 1; int px = k & 1;
  int hp = p >> 5; int wp = p & 31;
  pat[i] = f2bf(xf[((b << 6) + c) * 4096 + (hp * 2 + py) * 64 + (wp * 2 + px)]);
}

__global__ void k_bias_pos(const float* __restrict__ hraw, const float* __restrict__ pb,
                           const float* __restrict__ pos, bf16* __restrict__ hbf) {
  int i = blockIdx.x * 256 + threadIdx.x;      // 16384*192
  int e = i % 192; int row = i / 192; int p = row & 1023;
  hbf[i] = f2bf(hraw[i] + pb[e] + pos[p * 192 + e]);
}

// xc[(b*384+d)*1024 + l] = silu(causal_conv1d(x)), direction handled via lr
__global__ void k_conv1d(const float* __restrict__ xz, const float* __restrict__ wcv,
                         const float* __restrict__ bcv, float* __restrict__ xc, int dir) {
  int i = blockIdx.x * 256 + threadIdx.x;      // 16*384*1024
  int l = i & 1023; int bd = i >> 10; int d = bd % 384; int b = bd / 384;
  float acc = bcv[d];
#pragma unroll
  for (int t = 0; t < 4; ++t) {
    int ls = l - 3 + t;
    if (ls >= 0) {
      int lr = dir ? 1023 - ls : ls;
      acc += wcv[d * 4 + t] * xz[((b << 10) + lr) * 768 + d];
    }
  }
  xc[i] = siluf(acc);
}

// per (b,l): x_dbl = x_proj_w @ xc ; dt = dt_proj_w @ x_dbl[:12]; B,C split
__global__ void k_xproj_dt(const float* __restrict__ xc, const float* __restrict__ xpw,
                           const float* __restrict__ dtw, float* __restrict__ dt,
                           float* __restrict__ Bc, float* __restrict__ Cc) {
  __shared__ float xcs[384];
  __shared__ float xdbl[44];
  int row = blockIdx.x;                         // b*1024 + l
  int b = row >> 10; int l = row & 1023; int tid = threadIdx.x;
  for (int d = tid; d < 384; d += 64) xcs[d] = xc[((b * 384 + d) << 10) + l];
  __syncthreads();
  if (tid < 44) {
    float a = 0.f;
    for (int d = 0; d < 384; ++d) a += xpw[tid * 384 + d] * xcs[d];
    xdbl[tid] = a;
  }
  __syncthreads();
  for (int d = tid; d < 384; d += 64) {
    float a = 0.f;
#pragma unroll
    for (int r = 0; r < 12; ++r) a += xdbl[r] * dtw[d * 12 + r];
    dt[((b * 384 + d) << 10) + l] = a;
  }
  if (tid < 16)      Bc[row * 16 + tid]        = xdbl[12 + tid];
  else if (tid < 32) Cc[row * 16 + (tid - 16)] = xdbl[28 + (tid - 16)];
}

// selective scan: one thread per (b,d), 16 states in registers, L serial steps
__global__ void k_scan(const float* __restrict__ xc, const float* __restrict__ dt,
                       const float* __restrict__ Bc, const float* __restrict__ Cc,
                       const float* __restrict__ Alog, const float* __restrict__ Dp,
                       const float* __restrict__ dtb, const float* __restrict__ xz,
                       float* __restrict__ yacc, int dir) {
  int i = blockIdx.x * 256 + threadIdx.x;      // 6144 = 16*384
  int d = i % 384; int b = i / 384;
  float Ar[16], hs[16];
#pragma unroll
  for (int n = 0; n < 16; ++n) { Ar[n] = -__expf(Alog[d * 16 + n]); hs[n] = 0.f; }
  float Dd = Dp[d], bias = dtb[d];
  const float* dtp = dt + ((b * 384 + d) << 10);
  const float* xcp = xc + ((b * 384 + d) << 10);
  for (int t = 0; t < 1024; ++t) {
    float xv = dtp[t] + bias;
    float delta = xv > 20.f ? xv : log1pf(__expf(xv));
    float u = xcp[t];
    const float* Bt = Bc + (((b << 10) + t) << 4);
    const float* Ct = Cc + (((b << 10) + t) << 4);
    float y = 0.f;
#pragma unroll
    for (int n = 0; n < 16; ++n) {
      hs[n] = hs[n] * __expf(delta * Ar[n]) + delta * Bt[n] * u;
      y += hs[n] * Ct[n];
    }
    y += Dd * u;
    int lo = dir ? 1023 - t : t;
    float z = xz[((b << 10) + lo) * 768 + 384 + d];
    float ys = y * siluf(z);
    long idx = (long)((b << 10) + lo) * 384 + d;
    if (dir) yacc[idx] += ys; else yacc[idx] = ys;
  }
}

__global__ void k_mean(const float* __restrict__ outble, float* __restrict__ xfea) {
  int i = blockIdx.x * 256 + threadIdx.x;      // 3072 = 16*192
  int b = i / 192; int e = i % 192;
  float a = 0.f;
  for (int l = 0; l < 1024; ++l) a += outble[((b << 10) + l) * 192 + e];
  xfea[i] = a * (1.f / 1024.f);
}

__global__ void k_head(const float* __restrict__ xfea, const float* __restrict__ lng,
                       const float* __restrict__ lnb, const float* __restrict__ cw,
                       const float* __restrict__ cb, float* __restrict__ xcls) {
  __shared__ float s[192];
  __shared__ float ln[192];
  __shared__ float mu, ivar;
  int b = blockIdx.x; int e = threadIdx.x;
  float f = xfea[b * 192 + e];
  s[e] = f;
  __syncthreads();
  if (e == 0) {
    float m = 0.f;
    for (int j = 0; j < 192; ++j) m += s[j];
    m *= (1.f / 192.f);
    float v = 0.f;
    for (int j = 0; j < 192; ++j) { float dd = s[j] - m; v += dd * dd; }
    v *= (1.f / 192.f);
    mu = m; ivar = rsqrtf(v + EPSV);
  }
  __syncthreads();
  ln[e] = (f - mu) * ivar * lng[e] + lnb[e];
  __syncthreads();
  if (e < 16) {
    float a = cb[e];
    for (int j = 0; j < 192; ++j) a += cw[e * 192 + j] * ln[j];
    xcls[b * 16 + e] = a;
  }
}

// ---------------------------------------------------------------------------

enum {
  IX = 0, I_CONV1W, I_BN2G, I_BN2B, I_BN2RM, I_BN2RV, I_CONV2W,
  I_BN3G, I_BN3B, I_BN3RM, I_BN3RV, I_CONV3W, I_PROJW, I_PROJB, I_POS,
  I_INPROJW,
  IF_C1W, IF_C1B, IF_XPW, IF_DTW, IF_DTB, IF_ALOG, IF_D,
  IB_C1W, IB_C1B, IB_XPW, IB_DTW, IB_DTB, IB_ALOG, IB_D,
  I_OUTPROJW, I_LNG, I_LNB, I_CLSW, I_CLSB
};

extern "C" void kernel_launch(void* const* d_in, const int* in_sizes, int n_in,
                              void* d_out, int out_size, void* d_ws, size_t ws_size,
                              hipStream_t stream) {
  (void)in_sizes; (void)n_in; (void)out_size; (void)ws_size;
  const float* F[35];
  for (int i = 0; i < 35; ++i) F[i] = (const float*)d_in[i];
  float* out = (float*)d_out;

  char* base = (char*)d_ws;
  size_t off = 0;
  auto alloc = [&](size_t bytes) -> char* {
    char* p = base + off;
    off += (bytes + 255) & ~(size_t)255;
    return p;
  };

  // persistent region
  float* SUM   = (float*)alloc(4);
  bf16* wC1    = (bf16*)alloc((size_t)192 * 64 * 2);
  bf16* wC3    = (bf16*)alloc((size_t)64 * 576 * 2);
  bf16* wPj    = (bf16*)alloc((size_t)192 * 256 * 2);
  bf16* wIn    = (bf16*)alloc((size_t)768 * 192 * 2);
  bf16* wOut   = (bf16*)alloc((size_t)192 * 384 * 2);
  float* xf    = (float*)alloc((size_t)BATCH * CHN * HW * 4);
  size_t mark = off;

  // phase A scratch
  float* x0    = (float*)alloc((size_t)EMB * NCOL * 4);
  float* wfz   = (float*)alloc((size_t)EMB * NCOL * 4);
  float* x1    = (float*)alloc((size_t)EMB * NCOL * 4);
  bf16*  x2t   = (bf16*)alloc((size_t)NCOL * 576 * 2);
  float* c3o   = (float*)alloc((size_t)CHN * NCOL * 4);
  bf16*  xbt   = (bf16*)alloc((size_t)NCOL * 64 * 2);

  dim3 wave(32);

  k_zero1<<<1, 1, 0, stream>>>(SUM);
  k_cast_bf16<<<(192*64)/256, 256, 0, stream>>>(F[I_CONV1W], wC1, 192*64);
  k_cast_bf16<<<(64*576)/256, 256, 0, stream>>>(F[I_CONV3W], wC3, 64*576);
  k_cast_bf16<<<(192*256)/256, 256, 0, stream>>>(F[I_PROJW], wPj, 192*256);
  k_cast_bf16<<<(768*192)/256, 256, 0, stream>>>(F[I_INPROJW], wIn, 768*192);
  k_cast_bf16<<<(192*384)/256, 256, 0, stream>>>(F[I_OUTPROJW], wOut, 192*384);

  // conv1 (1x1) as WMMA GEMM: x0[192 x 65536]
  k_pack_x_bt<<<(64*NCOL)/256, 256, 0, stream>>>(F[IX], xbt);
  gemm_bf16_wmma<<<dim3(192/32, NCOL/16), wave, 0, stream>>>(wC1, xbt, x0, 192, NCOL, 64);

  // fuzzy weight (global reduction) + BN2/ReLU + depthwise 3x3
  k_fuzzy_sum<<<(EMB*NCOL)/256, 256, 0, stream>>>(x0, SUM);
  k_fuzzy_w<<<(EMB*NCOL)/256, 256, 0, stream>>>(x0, SUM, wfz);
  k_bnrelu_dwconv<<<(EMB*NCOL)/256, 256, 0, stream>>>(
      x0, F[I_BN2G], F[I_BN2B], F[I_BN2RM], F[I_BN2RV], F[I_CONV2W], x1);

  // concat + BN3/ReLU packed for conv3 GEMM, then conv3 + residual
  k_build_x2<<<(576*NCOL)/256, 256, 0, stream>>>(
      x0, wfz, x1, F[I_BN3G], F[I_BN3B], F[I_BN3RM], F[I_BN3RV], x2t);
  gemm_bf16_wmma<<<dim3(64/32, NCOL/16), wave, 0, stream>>>(wC3, x2t, c3o, 64, NCOL, 576);
  k_residual<<<(BATCH*CHN*HW)/256, 256, 0, stream>>>(c3o, F[IX], xf);

  // phase B reuses phase-A scratch
  off = mark;
  bf16*  pat   = (bf16*)alloc((size_t)NROW * PDIM * 2);
  float* hraw  = (float*)alloc((size_t)NROW * EMB * 4);
  bf16*  hbf   = (bf16*)alloc((size_t)NROW * EMB * 2);
  float* xz    = (float*)alloc((size_t)NROW * 768 * 4);
  float* xc    = (float*)alloc((size_t)BATCH * DINNER * SEQL * 4);
  float* dt    = (float*)alloc((size_t)BATCH * DINNER * SEQL * 4);
  float* Bc    = (float*)alloc((size_t)BATCH * SEQL * DSTATE * 4);
  float* Cc    = (float*)alloc((size_t)BATCH * SEQL * DSTATE * 4);
  float* yacc  = (float*)alloc((size_t)NROW * DINNER * 4);
  bf16*  ybf   = (bf16*)alloc((size_t)NROW * DINNER * 2);
  float* oble  = (float*)alloc((size_t)NROW * EMB * 4);

  // patch embed
  k_pack_patches<<<(NROW*PDIM)/256, 256, 0, stream>>>(xf, pat);
  gemm_bf16_wmma<<<dim3(NROW/32, EMB/16), wave, 0, stream>>>(pat, wPj, hraw, NROW, EMB, PDIM);
  k_bias_pos<<<(NROW*EMB)/256, 256, 0, stream>>>(hraw, F[I_PROJB], F[I_POS], hbf);

  // in_proj
  gemm_bf16_wmma<<<dim3(NROW/32, 768/16), wave, 0, stream>>>(hbf, wIn, xz, NROW, 768, EMB);

  // forward direction
  k_conv1d<<<(BATCH*DINNER*SEQL)/256, 256, 0, stream>>>(xz, F[IF_C1W], F[IF_C1B], xc, 0);
  k_xproj_dt<<<NROW, 64, 0, stream>>>(xc, F[IF_XPW], F[IF_DTW], dt, Bc, Cc);
  k_scan<<<(BATCH*DINNER)/256, 256, 0, stream>>>(
      xc, dt, Bc, Cc, F[IF_ALOG], F[IF_D], F[IF_DTB], xz, yacc, 0);

  // backward direction (accumulates)
  k_conv1d<<<(BATCH*DINNER*SEQL)/256, 256, 0, stream>>>(xz, F[IB_C1W], F[IB_C1B], xc, 1);
  k_xproj_dt<<<NROW, 64, 0, stream>>>(xc, F[IB_XPW], F[IB_DTW], dt, Bc, Cc);
  k_scan<<<(BATCH*DINNER)/256, 256, 0, stream>>>(
      xc, dt, Bc, Cc, F[IB_ALOG], F[IB_D], F[IB_DTB], xz, yacc, 1);

  // out_proj
  k_cast_bf16<<<(NROW*DINNER)/256, 256, 0, stream>>>(yacc, ybf, NROW*DINNER);
  gemm_bf16_wmma<<<dim3(NROW/32, EMB/16), wave, 0, stream>>>(ybf, wOut, oble, NROW, EMB, DINNER);

  // mean + LN + classifier -> d_out = [x_fea (16x192) | x_cls (16x16)]
  k_mean<<<(BATCH*EMB)/256, 256, 0, stream>>>(oble, out);
  k_head<<<BATCH, 192, 0, stream>>>(out, F[I_LNG], F[I_LNB], F[I_CLSW], F[I_CLSB],
                                    out + BATCH * EMB);
}
